// ComplexCrossAttention_81647328297520
// MI455X (gfx1250) — matrix operand order repaired and checked by reading, not compiled, not run on hardware
//
#include <hip/hip_runtime.h>
#include <hip/hip_bf16.h>
#include <math.h>

// ---------------------------------------------------------------------------
// CDNA5 (gfx1250) complex transformer block.
// GEMMs: bf16 hi/lo split -> v_wmma_f32_16x16x32_bf16 (3 WMMA per f32 product)
// Attention: exact f32 -> v_wmma_f32_16x16x4_f32
// ---------------------------------------------------------------------------

typedef __attribute__((ext_vector_type(16))) __bf16 v16bf;
typedef __attribute__((ext_vector_type(8)))  float  v8f;
typedef __attribute__((ext_vector_type(2)))  float  v2f;

#define DEVINL static __device__ __forceinline__

DEVINL unsigned short f32_bf16(float x) {
  union { float f; unsigned u; } c; c.f = x;
  unsigned r = c.u + 0x7fffu + ((c.u >> 16) & 1u);   // round-to-nearest-even
  return (unsigned short)(r >> 16);
}
DEVINL float bf16_f32(unsigned short s) {
  union { unsigned u; float f; } c; c.u = ((unsigned)s) << 16;
  return c.f;
}

union BF16Frag { v16bf v; uint4 q[2]; };

DEVINL v8f wmma_bf16(const BF16Frag& a, const BF16Frag& b, v8f c) {
  return __builtin_amdgcn_wmma_f32_16x16x32_bf16(false, a.v, false, b.v,
                                                 (short)0, c, false, false);
}
DEVINL v8f wmma_f32(v2f a, v2f b, v8f c) {
  return __builtin_amdgcn_wmma_f32_16x16x4_f32(false, a, false, b,
                                               (short)0, c, false, false);
}

#define Z8 {0.f,0.f,0.f,0.f,0.f,0.f,0.f,0.f}

// ---------------------------------------------------------------------------
// Complex GEMM:  Y = X @ W^T + bias   (all complex)
//   X: [M x K] via Xr/Xi with (rowStride, elemStride)   (supports interleaved)
//   W: [N x K x 2] interleaved (real at +0, imag at +1)
//   Y: planes Yr, Yi row-major [M x N]
// ACT==1: modReLU epilogue  hr=0.5*(hr+|h|), hi=0.5*hi
// Block: 256 thr (8 waves), tile 128x64, k-step 32, bf16 hi/lo split.
// ---------------------------------------------------------------------------
#define BM 128
#define BN 64
#define BK 32
#define APAD 40   // row stride in ushorts (80B: 16B aligned, conflict-skewed)

template<int ACT>
__global__ __launch_bounds__(256)
void cgemm_bf16x3(const float* __restrict__ Xr, const float* __restrict__ Xi,
                  int xRowStride, int xElemStride,
                  const float* __restrict__ W, const float* __restrict__ Bias,
                  int Kdim, int Ndim,
                  float* __restrict__ Yr, float* __restrict__ Yi)
{
  __shared__ __align__(16) unsigned short Arh[BM][APAD], Arl[BM][APAD];
  __shared__ __align__(16) unsigned short Aih[BM][APAD], Ail[BM][APAD];
  __shared__ __align__(16) unsigned short Brh[BN][APAD], Brl[BN][APAD];
  __shared__ __align__(16) unsigned short Bih[BN][APAD], Bil[BN][APAD];

  const int tid  = threadIdx.x;
  const int lane = tid & 31, wave = tid >> 5;
  const int wm = wave >> 1, wn = wave & 1;          // 4x2 wave grid
  const int gm0 = blockIdx.y * BM, gn0 = blockIdx.x * BN;
  const int lh = lane & 15, g = lane >> 4;

  // acc[msub][nsub][p]: p0=xr*wr p1=xi*wi p2=xr*wi p3=xi*wr
  v8f acc[2][2][4];
#pragma unroll
  for (int a = 0; a < 2; a++)
#pragma unroll
    for (int b = 0; b < 2; b++)
#pragma unroll
      for (int p = 0; p < 4; p++) acc[a][b][p] = (v8f)Z8;

  for (int k0 = 0; k0 < Kdim; k0 += BK) {
    __syncthreads();
    // ---- stage A panel: 128 x 32 (xr, xi) as bf16 hi/lo
#pragma unroll
    for (int it = 0; it < (BM * BK) / 256; it++) {
      int idx = tid + it * 256;
      int r = idx >> 5, k = idx & 31;
      long o = (long)(gm0 + r) * xRowStride + (long)(k0 + k) * xElemStride;
      float vr = Xr[o], vi = Xi[o];
      unsigned short h;
      h = f32_bf16(vr); Arh[r][k] = h; Arl[r][k] = f32_bf16(vr - bf16_f32(h));
      h = f32_bf16(vi); Aih[r][k] = h; Ail[r][k] = f32_bf16(vi - bf16_f32(h));
    }
    // ---- stage B panel: 64 x 32 (wr, wi) as bf16 hi/lo
#pragma unroll
    for (int it = 0; it < (BN * BK) / 256; it++) {
      int idx = tid + it * 256;
      int n = idx >> 5, k = idx & 31;
      long o = ((long)(gn0 + n) * Kdim + (k0 + k)) * 2;
      float wr = W[o], wi = W[o + 1];
      unsigned short h;
      h = f32_bf16(wr); Brh[n][k] = h; Brl[n][k] = f32_bf16(wr - bf16_f32(h));
      h = f32_bf16(wi); Bih[n][k] = h; Bil[n][k] = f32_bf16(wi - bf16_f32(h));
    }
    // prefetch next K panels into GL2
    if (k0 + BK < Kdim) {
      __builtin_prefetch(&Xr[(long)(gm0 + (tid >> 1)) * xRowStride +
                             (long)(k0 + BK) * xElemStride], 0, 1);
      __builtin_prefetch(&W[((long)(gn0 + (tid & 63)) * Kdim + (k0 + BK)) * 2], 0, 1);
    }
    __syncthreads();

    // ---- B fragments (wr/wi hi/lo for both n-subtiles)
    BF16Frag bRh[2], bRl[2], bIh[2], bIl[2];
#pragma unroll
    for (int ns = 0; ns < 2; ns++) {
      int nr = wn * 32 + ns * 16 + lh;
      bRh[ns].q[0] = *reinterpret_cast<const uint4*>(&Brh[nr][g * 8]);
      bRh[ns].q[1] = *reinterpret_cast<const uint4*>(&Brh[nr][16 + g * 8]);
      bRl[ns].q[0] = *reinterpret_cast<const uint4*>(&Brl[nr][g * 8]);
      bRl[ns].q[1] = *reinterpret_cast<const uint4*>(&Brl[nr][16 + g * 8]);
      bIh[ns].q[0] = *reinterpret_cast<const uint4*>(&Bih[nr][g * 8]);
      bIh[ns].q[1] = *reinterpret_cast<const uint4*>(&Bih[nr][16 + g * 8]);
      bIl[ns].q[0] = *reinterpret_cast<const uint4*>(&Bil[nr][g * 8]);
      bIl[ns].q[1] = *reinterpret_cast<const uint4*>(&Bil[nr][16 + g * 8]);
    }
    // ---- group 1: xr * {wr,wi}
    {
      BF16Frag aH[2], aL[2];
#pragma unroll
      for (int ms = 0; ms < 2; ms++) {
        int mr = wm * 32 + ms * 16 + lh;
        aH[ms].q[0] = *reinterpret_cast<const uint4*>(&Arh[mr][g * 8]);
        aH[ms].q[1] = *reinterpret_cast<const uint4*>(&Arh[mr][16 + g * 8]);
        aL[ms].q[0] = *reinterpret_cast<const uint4*>(&Arl[mr][g * 8]);
        aL[ms].q[1] = *reinterpret_cast<const uint4*>(&Arl[mr][16 + g * 8]);
      }
#pragma unroll
      for (int ms = 0; ms < 2; ms++)
#pragma unroll
        for (int ns = 0; ns < 2; ns++) {
          v8f c0 = acc[ms][ns][0];
          c0 = wmma_bf16(aH[ms], bRh[ns], c0);
          c0 = wmma_bf16(aH[ms], bRl[ns], c0);
          c0 = wmma_bf16(aL[ms], bRh[ns], c0);
          acc[ms][ns][0] = c0;
          v8f c2 = acc[ms][ns][2];
          c2 = wmma_bf16(aH[ms], bIh[ns], c2);
          c2 = wmma_bf16(aH[ms], bIl[ns], c2);
          c2 = wmma_bf16(aL[ms], bIh[ns], c2);
          acc[ms][ns][2] = c2;
        }
    }
    // ---- group 2: xi * {wi,wr}
    {
      BF16Frag aH[2], aL[2];
#pragma unroll
      for (int ms = 0; ms < 2; ms++) {
        int mr = wm * 32 + ms * 16 + lh;
        aH[ms].q[0] = *reinterpret_cast<const uint4*>(&Aih[mr][g * 8]);
        aH[ms].q[1] = *reinterpret_cast<const uint4*>(&Aih[mr][16 + g * 8]);
        aL[ms].q[0] = *reinterpret_cast<const uint4*>(&Ail[mr][g * 8]);
        aL[ms].q[1] = *reinterpret_cast<const uint4*>(&Ail[mr][16 + g * 8]);
      }
#pragma unroll
      for (int ms = 0; ms < 2; ms++)
#pragma unroll
        for (int ns = 0; ns < 2; ns++) {
          v8f c1 = acc[ms][ns][1];
          c1 = wmma_bf16(aH[ms], bIh[ns], c1);
          c1 = wmma_bf16(aH[ms], bIl[ns], c1);
          c1 = wmma_bf16(aL[ms], bIh[ns], c1);
          acc[ms][ns][1] = c1;
          v8f c3 = acc[ms][ns][3];
          c3 = wmma_bf16(aH[ms], bRh[ns], c3);
          c3 = wmma_bf16(aH[ms], bRl[ns], c3);
          c3 = wmma_bf16(aL[ms], bRh[ns], c3);
          acc[ms][ns][3] = c3;
        }
    }
  }

  // ---- epilogue: complex combine + bias (+ modReLU), C layout M=j+8*g, N=lh
#pragma unroll
  for (int ms = 0; ms < 2; ms++)
#pragma unroll
    for (int ns = 0; ns < 2; ns++) {
      int n = gn0 + wn * 32 + ns * 16 + lh;
      float br = Bias[2 * n], bi = Bias[2 * n + 1];
#pragma unroll
      for (int j = 0; j < 8; j++) {
        int m = gm0 + wm * 32 + ms * 16 + j + g * 8;
        float yr = acc[ms][ns][0][j] - acc[ms][ns][1][j] + br;
        float yi = acc[ms][ns][2][j] + acc[ms][ns][3][j] + bi;
        if (ACT == 1) {
          float mag = sqrtf(yr * yr + yi * yi);
          yr = 0.5f * (yr + mag);
          yi = 0.5f * yi;
        }
        long o = (long)m * Ndim + n;
        Yr[o] = yr; Yi[o] = yi;
      }
    }
}

// ---------------------------------------------------------------------------
// Complex attention for one (b, h, 32-query tile). 256 threads / 8 waves.
// Scores in LDS, K/V streamed in 64-key chunks, exact f32 WMMA (16x16x4).
// ---------------------------------------------------------------------------
__global__ __launch_bounds__(256)
void cattn(const float* __restrict__ Qr, const float* __restrict__ Qi,
           const float* __restrict__ Kr, const float* __restrict__ Ki,
           const float* __restrict__ Vr, const float* __restrict__ Vi,
           float* __restrict__ Or_, float* __restrict__ Oi_)
{
  __shared__ float Sr[32][514], Si[32][514];   // scores / probs
  __shared__ float Tile[64][66];               // K (transposed) / V staging

  const int tid = threadIdx.x, lane = tid & 31, wave = tid >> 5;
  const int lh = lane & 15, g = lane >> 4;
  const int q0 = blockIdx.x * 32, h = blockIdx.y, b = blockIdx.z;
  const long rowBase = ((long)b * 512) * 1024 + (long)h * 64;
  const float scale = 0.125f;                  // 1/sqrt(64)

  const int qsub = wave & 1;                   // q subtile (phase A & B)
  const int ksub = wave >> 1;                  // key subtile (phase A), 0..3
  const int dsub = wave >> 1;                  // dh subtile (phase B), 0..3

  // Q fragments held in registers for the whole kernel: 16 k-steps x (r,i)
  v2f aQr[16], aQi[16];
  {
    long qrow = rowBase + (long)(q0 + qsub * 16 + lh) * 1024;
#pragma unroll
    for (int s = 0; s < 16; s++) {
      int k = s * 4 + g * 2;
      aQr[s].x = Qr[qrow + k]; aQr[s].y = Qr[qrow + k + 1];
      aQi[s].x = Qi[qrow + k]; aQi[s].y = Qi[qrow + k + 1];
    }
  }

  // ---- phase A: scores  Sr = (QrKr - QiKi)*s,  Si = (QiKr + QrKi)*s
  for (int kc = 0; kc < 8; kc++) {
    int key0 = kc * 64;
    for (int pass = 0; pass < 2; pass++) {     // pass0: Kr, pass1: Ki
      const float* Ksrc = pass ? Ki : Kr;
      __syncthreads();
      for (int idx = tid; idx < 64 * 64; idx += 256) {
        int d = idx & 63, ky = idx >> 6;
        Tile[d][ky] = Ksrc[rowBase + (long)(key0 + ky) * 1024 + d];  // K^T
      }
      __syncthreads();
      v8f a0 = (v8f)Z8, a1 = (v8f)Z8;
#pragma unroll
      for (int s = 0; s < 16; s++) {
        int k = s * 4 + g * 2;
        v2f bf; bf.x = Tile[k][ksub * 16 + lh]; bf.y = Tile[k + 1][ksub * 16 + lh];
        a0 = wmma_f32(aQr[s], bf, a0);
        a1 = wmma_f32(aQi[s], bf, a1);
      }
      int col = key0 + ksub * 16 + lh;
#pragma unroll
      for (int j = 0; j < 8; j++) {
        int row = qsub * 16 + j + g * 8;
        if (pass == 0) { Sr[row][col]  = a0[j] * scale; Si[row][col]  = a1[j] * scale; }
        else           { Sr[row][col] -= a1[j] * scale; Si[row][col] += a0[j] * scale; }
      }
    }
  }
  __syncthreads();

  // ---- softmax over keys, independently for Sr and Si (stable)
  for (int rr = 0; rr < 4; rr++) {
    int row = wave * 4 + rr;
    for (int comp = 0; comp < 2; comp++) {
      float (*S)[514] = comp ? Si : Sr;
      float v[16], m = -3.0e38f;
#pragma unroll
      for (int t = 0; t < 16; t++) { v[t] = S[row][lane + t * 32]; m = fmaxf(m, v[t]); }
      for (int o = 16; o; o >>= 1) m = fmaxf(m, __shfl_xor(m, o, 32));
      float sum = 0.f;
#pragma unroll
      for (int t = 0; t < 16; t++) { v[t] = expf(v[t] - m); sum += v[t]; }
      for (int o = 16; o; o >>= 1) sum += __shfl_xor(sum, o, 32);
      float inv = 1.f / sum;
#pragma unroll
      for (int t = 0; t < 16; t++) S[row][lane + t * 32] = v[t] * inv;
    }
  }
  __syncthreads();

  // ---- phase B: O = P @ V (complex)
  v8f o0 = (v8f)Z8, o1 = (v8f)Z8, o2 = (v8f)Z8, o3 = (v8f)Z8;
  for (int kc = 0; kc < 8; kc++) {
    int key0 = kc * 64;
    for (int pass = 0; pass < 2; pass++) {     // pass0: Vr, pass1: Vi
      const float* Vsrc = pass ? Vi : Vr;
      __syncthreads();
      for (int idx = tid; idx < 64 * 64; idx += 256) {
        int d = idx & 63, ky = idx >> 6;
        Tile[ky][d] = Vsrc[rowBase + (long)(key0 + ky) * 1024 + d];
      }
      __syncthreads();
      int prow = qsub * 16 + lh;
#pragma unroll
      for (int s = 0; s < 16; s++) {
        int k = s * 4 + g * 2;
        int kg = key0 + k;
        v2f ar, ai, bf;
        ar.x = Sr[prow][kg]; ar.y = Sr[prow][kg + 1];
        ai.x = Si[prow][kg]; ai.y = Si[prow][kg + 1];
        bf.x = Tile[k][dsub * 16 + lh]; bf.y = Tile[k + 1][dsub * 16 + lh];
        if (pass == 0) { o0 = wmma_f32(ar, bf, o0); o3 = wmma_f32(ai, bf, o3); }
        else           { o2 = wmma_f32(ar, bf, o2); o1 = wmma_f32(ai, bf, o1); }
      }
    }
  }
  // Or = ArVr - AiVi ; Oi = ArVi + AiVr
  int n = dsub * 16 + lh;
#pragma unroll
  for (int j = 0; j < 8; j++) {
    int q = q0 + qsub * 16 + j + g * 8;
    long o = rowBase + (long)q * 1024 + n;
    Or_[o] = o0[j] - o1[j];
    Oi_[o] = o2[j] + o3[j];
  }
}

// ---------------------------------------------------------------------------
// Fused residual + LayerNorm over D=1024.  grid = (B*S, 2 comps), 256 thr.
// element addr = base_ptr[(row*1024 + d) * stride]
// ---------------------------------------------------------------------------
__global__ __launch_bounds__(256)
void ln_fused(const float* __restrict__ a_r, const float* __restrict__ a_i, int aStride,
              const float* __restrict__ r_r, const float* __restrict__ r_i, int rStride,
              const float* __restrict__ g_base, const float* __restrict__ b_base,
              float* __restrict__ o_r, float* __restrict__ o_i, int oStride)
{
  const int row = blockIdx.x, comp = blockIdx.y, tid = threadIdx.x;
  const float* A = comp ? a_i : a_r;
  const float* R = comp ? r_i : r_r;
  float*       O = comp ? o_i : o_r;
  const float* G = g_base + comp * 1024;
  const float* Bb = b_base + comp * 1024;
  __shared__ float red[256];
  const long base = (long)row * 1024;

  float v[4];
#pragma unroll
  for (int t = 0; t < 4; t++) {
    long d = tid + t * 256;
    v[t] = A[(base + d) * aStride] + R[(base + d) * rStride];
  }
  float s = v[0] + v[1] + v[2] + v[3];
  red[tid] = s; __syncthreads();
  for (int o = 128; o; o >>= 1) { if (tid < o) red[tid] += red[tid + o]; __syncthreads(); }
  float mean = red[0] * (1.f / 1024.f);
  __syncthreads();
  float sq = 0.f;
#pragma unroll
  for (int t = 0; t < 4; t++) { float d = v[t] - mean; sq += d * d; }
  red[tid] = sq; __syncthreads();
  for (int o = 128; o; o >>= 1) { if (tid < o) red[tid] += red[tid + o]; __syncthreads(); }
  float rstd = rsqrtf(red[0] * (1.f / 1024.f) + 1e-5f);
#pragma unroll
  for (int t = 0; t < 4; t++) {
    long d = tid + t * 256;
    O[(base + d) * oStride] = (v[t] - mean) * rstd * G[d] + Bb[d];
  }
}

// ---------------------------------------------------------------------------
extern "C" void kernel_launch(void* const* d_in, const int* in_sizes, int n_in,
                              void* d_out, int out_size, void* d_ws, size_t ws_size,
                              hipStream_t stream) {
  const float* x     = (const float*)d_in[0];
  const float* query = (const float*)d_in[1];
  const float* wq    = (const float*)d_in[2];
  const float* bq    = (const float*)d_in[3];
  const float* wk    = (const float*)d_in[4];
  const float* bk    = (const float*)d_in[5];
  const float* wv    = (const float*)d_in[6];
  const float* bv    = (const float*)d_in[7];
  const float* wfc   = (const float*)d_in[8];
  const float* bfc   = (const float*)d_in[9];
  const float* wpj   = (const float*)d_in[10];
  const float* bpj   = (const float*)d_in[11];
  const float* lng   = (const float*)d_in[12];
  const float* lnb   = (const float*)d_in[13];
  float* out = (float*)d_out;
  float* wsf = (float*)d_ws;

  const long P = 4096L * 1024L;                // one plane
  float *Qr = wsf,        *Qi = wsf + P;
  float *Kr = wsf + 2*P,  *Ki = wsf + 3*P;
  float *Vr = wsf + 4*P,  *Vi = wsf + 5*P;
  float *Or_ = wsf + 6*P, *Oi_ = wsf + 7*P;
  float *Xr2 = wsf + 8*P, *Xi2 = wsf + 9*P;
  float *Mr = wsf + 10*P, *Mi = wsf + 11*P;
  float *Hr = wsf;                             // 4096x4096, aliases dead Q..Ki
  float *Hi = wsf + 4*P;                       // aliases dead V..O

  dim3 blk(256);

  // complex QKV projections (Q from query; K,V from x)
  cgemm_bf16x3<0><<<dim3(1024/BN, 4096/BM), blk, 0, stream>>>(
      query, query + 1, 2048, 2, wq, bq, 1024, 1024, Qr, Qi);
  cgemm_bf16x3<0><<<dim3(1024/BN, 4096/BM), blk, 0, stream>>>(
      x, x + 1, 2048, 2, wk, bk, 1024, 1024, Kr, Ki);
  cgemm_bf16x3<0><<<dim3(1024/BN, 4096/BM), blk, 0, stream>>>(
      x, x + 1, 2048, 2, wv, bv, 1024, 1024, Vr, Vi);

  // complex multi-head attention
  cattn<<<dim3(16, 16, 8), blk, 0, stream>>>(Qr, Qi, Kr, Ki, Vr, Vi, Or_, Oi_);

  // O' = LN(O + query)  (in-place), then x2 = LN(x + O')
  ln_fused<<<dim3(4096, 2), blk, 0, stream>>>(
      Or_, Oi_, 1, query, query + 1, 2, lng, lnb, Or_, Oi_, 1);
  ln_fused<<<dim3(4096, 2), blk, 0, stream>>>(
      Or_, Oi_, 1, x, x + 1, 2, lng + 2048, lnb + 2048, Xr2, Xi2, 1);

  // complex MLP: fc + modReLU, then proj
  cgemm_bf16x3<1><<<dim3(4096/BN, 4096/BM), blk, 0, stream>>>(
      Xr2, Xi2, 1024, 1, wfc, bfc, 1024, 4096, Hr, Hi);
  cgemm_bf16x3<0><<<dim3(1024/BN, 4096/BM), blk, 0, stream>>>(
      Hr, Hi, 4096, 1, wpj, bpj, 4096, 1024, Mr, Mi);

  // y = LN(x2 + m) -> interleaved [B,S,D,2] output
  ln_fused<<<dim3(4096, 2), blk, 0, stream>>>(
      Mr, Mi, 1, Xr2, Xi2, 1, lng + 4096, lnb + 4096, out, out + 1, 2);
}
